// S3Mamba_Pan_38465727103458
// MI455X (gfx1250) — compile-verified, matrix operand order checked by reading, not run on hardware
//
#include <hip/hip_runtime.h>
#include <math.h>

typedef __attribute__((ext_vector_type(16))) _Float16 v16h;
typedef __attribute__((ext_vector_type(8)))  _Float16 v8h;
typedef __attribute__((ext_vector_type(8)))  float    v8f;
typedef unsigned int v4u __attribute__((ext_vector_type(4)));
typedef int          v8i __attribute__((ext_vector_type(8)));
typedef int          v4i __attribute__((ext_vector_type(4)));

// ---------------------------------------------------------------------------
// TDM helper: DMA a 64-row x 64-half tile of a [rows,K] f16 matrix into LDS,
// with hardware padding to a 72-half row stride and OOB zero-fill.
// Issued by one wave; completion tracked with TENSORcnt.
// ---------------------------------------------------------------------------
__device__ __forceinline__ void tdm_load_b_tile(const _Float16* base, unsigned lds_off,
                                                unsigned rows_left, unsigned k_left,
                                                int row0, int k0, int K)
{
  unsigned long long ga = (unsigned long long)(size_t)base +
                          ((unsigned long long)row0 * (unsigned)K + (unsigned)k0) * 2ull;
  v4u g0;
  g0[0] = 1u;                                   // count=1, user descriptor
  g0[1] = lds_off;                              // LDS byte address
  g0[2] = (unsigned)(ga & 0xffffffffu);
  g0[3] = (unsigned)((ga >> 32) & 0x01ffffffu) | (2u << 30);   // addr[56:32] | type=2
  v8i g1;
  unsigned td0 = k_left;                        // tensor_dim0 (elements in row)
  unsigned td1 = rows_left;                     // tensor_dim1 (rows)
  g1[0] = (int)((1u << 16)        // data_size = 2 bytes
              | (1u << 20)        // pad_enable
              | (4u << 22)        // pad_interval: every 32 dwords
              | (3u << 25));      // pad_amount: 4 dwords (64 -> 72 halves/row)
  g1[1] = (int)((td0 & 0xffffu) << 16);
  g1[2] = (int)((td0 >> 16) | ((td1 & 0xffffu) << 16));
  g1[3] = (int)((td1 >> 16) | (64u << 16));     // tile_dim0 = 64
  g1[4] = (int)(64u);                           // tile_dim1 = 64, tile_dim2 = 0
  g1[5] = (int)(unsigned)K;                     // tensor_dim0_stride (low 32)
  g1[6] = 0;
  g1[7] = 0;
  v4i z4 = {0, 0, 0, 0};
#if __clang_major__ >= 23
  v8i z8 = {0, 0, 0, 0, 0, 0, 0, 0};
  __builtin_amdgcn_tensor_load_to_lds(g0, g1, z4, z4, z8, 0);
#else
  __builtin_amdgcn_tensor_load_to_lds(g0, g1, z4, z4, 0);
#endif
}

// ---------------------------------------------------------------------------
// WMMA GEMM:  Y[M,N] = act( X[M,K](row stride lda) @ W16[N,K]^T + bias[N] )
// X f32 (converted to f16 while staging), W16 pre-converted f16 (TDM-staged),
// f32 accumulate.  Double-buffered LDS, one barrier per K-tile.
// act: 0=none, 1=silu
// ---------------------------------------------------------------------------
#define GTM 64
#define GTN 64
#define GTK 64
#define GTKP 72   // padded LDS row length in halves (144B, 16B aligned)

__global__ __launch_bounds__(128) void wmma_gemm_kernel(
    const float* __restrict__ X, int lda,
    const _Float16* __restrict__ W16,
    const float* __restrict__ bias,
    float* __restrict__ Y, int ldc,
    int M, int N, int K, int act)
{
  __shared__ _Float16 sA[2][GTM * GTKP];
  __shared__ _Float16 sB[2][GTN * GTKP];

  const int tid   = threadIdx.x;
  const int lane  = tid & 31;
  const int wave  = tid >> 5;
  const int waveM = wave >> 1;
  const int waveN = wave & 1;
  const int tileM = blockIdx.x * GTM;
  const int tileN = blockIdx.y * GTN;
  const int r = lane & 15;
  const int g = lane >> 4;

  v8f acc[2][2] = {};

  auto stageA = [&](int k0, int buf) {
    for (int i = tid; i < (GTM * GTK) / 4; i += 128) {
      int row  = i >> 4;
      int col4 = (i & 15) << 2;
      int gk = k0 + col4;
      float4 va = make_float4(0.f, 0.f, 0.f, 0.f);
      int ar = tileM + row;
      if (ar < M && gk < K) va = *(const float4*)(X + (size_t)ar * lda + gk);
      _Float16* pa = &sA[buf][row * GTKP + col4];
      pa[0] = (_Float16)va.x; pa[1] = (_Float16)va.y;
      pa[2] = (_Float16)va.z; pa[3] = (_Float16)va.w;
    }
  };
  auto stageB = [&](int k0, int buf) {
    if (wave == 0)
      tdm_load_b_tile(W16, (unsigned)(size_t)(void*)&sB[buf][0],
                      (unsigned)(N - tileN), (unsigned)(K - k0), tileN, k0, K);
  };

  const int nk = (K + GTK - 1) / GTK;
  stageA(0, 0);
  stageB(0, 0);

  for (int it = 0; it < nk; it++) {
    int cur = it & 1;
    int nxt = cur ^ 1;
    if (wave == 0) __builtin_amdgcn_s_wait_tensorcnt(0);  // publish TDM tile
    __syncthreads();

    if (it + 1 < nk) {       // overlap next-tile staging with current compute
      stageA((it + 1) * GTK, nxt);
      stageB((it + 1) * GTK, nxt);
      int prow = tileM + (tid >> 1);
      if (prow < M)
        __builtin_prefetch(X + (size_t)prow * lda + (it + 2) * GTK + (tid & 1) * 32, 0, 3);
    }

#pragma unroll
    for (int kk = 0; kk < GTK; kk += 32) {
      v16h a[2], b[2];
#pragma unroll
      for (int mi = 0; mi < 2; mi++) {
        const _Float16* p = &sA[cur][(waveM * 32 + mi * 16 + r) * GTKP + kk + 8 * g];
        v8h lo = *(const v8h*)p;
        v8h hi = *(const v8h*)(p + 16);
        a[mi] = __builtin_shufflevector(lo, hi, 0,1,2,3,4,5,6,7,8,9,10,11,12,13,14,15);
      }
#pragma unroll
      for (int ni = 0; ni < 2; ni++) {
        const _Float16* p = &sB[cur][(waveN * 32 + ni * 16 + r) * GTKP + kk + 8 * g];
        v8h lo = *(const v8h*)p;
        v8h hi = *(const v8h*)(p + 16);
        b[ni] = __builtin_shufflevector(lo, hi, 0,1,2,3,4,5,6,7,8,9,10,11,12,13,14,15);
      }
#pragma unroll
      for (int mi = 0; mi < 2; mi++)
#pragma unroll
        for (int ni = 0; ni < 2; ni++)
          acc[mi][ni] = __builtin_amdgcn_wmma_f32_16x16x32_f16(
              false, a[mi], false, b[ni], (short)0, acc[mi][ni], false, false);
    }
  }

  // Epilogue: C/D layout = lanes 0-15: N=lane, M=v; lanes 16-31: N=lane-16, M=v+8
#pragma unroll
  for (int mi = 0; mi < 2; mi++) {
#pragma unroll
    for (int ni = 0; ni < 2; ni++) {
      int col = tileN + waveN * 32 + ni * 16 + r;
      if (col >= N) continue;
      float bv = bias ? bias[col] : 0.f;
      int mbase = tileM + waveM * 32 + mi * 16 + 8 * g;
#pragma unroll
      for (int v = 0; v < 8; v++) {
        int row = mbase + v;
        if (row >= M) continue;
        float val = acc[mi][ni][v] + bv;
        if (act == 1) val = val / (1.f + __expf(-val));
        Y[(size_t)row * ldc + col] = val;
      }
    }
  }
}

// f32 -> f16 weight conversion
__global__ void f32_to_f16_kernel(const float* __restrict__ src, _Float16* __restrict__ dst, int n)
{
  int i = blockIdx.x * blockDim.x + threadIdx.x;
  if (i < n) dst[i] = (_Float16)src[i];
}

// ---------------------------------------------------------------------------
// Direct 3x3 conv, pad 1.  leaky: slope for x<0 (1.0 = identity). Optional
// residual add (same shape as output).
// ---------------------------------------------------------------------------
__global__ void conv3x3_kernel(const float* __restrict__ x, const float* __restrict__ w,
                               const float* __restrict__ bias, const float* __restrict__ residual,
                               float* __restrict__ y, int B, int Cin, int Cout,
                               int H, int W, float leaky)
{
  int co = blockIdx.x, b = blockIdx.y;
  int HW = H * W;
  const float* xb = x + (size_t)b * Cin * HW;
  float* yb = y + ((size_t)b * Cout + co) * HW;
  const float* wc = w + (size_t)co * Cin * 9;
  float bv = bias ? bias[co] : 0.f;
  for (int p = threadIdx.x; p < HW; p += blockDim.x) {
    int i = p / W, j = p % W;
    float acc = bv;
    for (int ci = 0; ci < Cin; ci++) {
      const float* xp = xb + (size_t)ci * HW;
      const float* wp = wc + ci * 9;
      for (int di = -1; di <= 1; di++) {
        int ii = i + di; if (ii < 0 || ii >= H) continue;
        for (int dj = -1; dj <= 1; dj++) {
          int jj = j + dj; if (jj < 0 || jj >= W) continue;
          acc += wp[(di + 1) * 3 + (dj + 1)] * xp[ii * W + jj];
        }
      }
    }
    if (acc < 0.f) acc *= leaky;
    if (residual) acc += residual[((size_t)b * Cout + co) * HW + p];
    yb[p] = acc;
  }
}

// global average pool: x (B,C,HW) -> out (B,C).  grid (C,B)
__global__ void gap_kernel(const float* __restrict__ x, float* __restrict__ out, int HW)
{
  int c = blockIdx.x, b = blockIdx.y, C = gridDim.x;
  const float* p = x + ((size_t)b * C + c) * HW;
  float s = 0.f;
  for (int i = threadIdx.x; i < HW; i += blockDim.x) s += p[i];
  __shared__ float red[256];
  red[threadIdx.x] = s; __syncthreads();
  for (int o = 128; o > 0; o >>= 1) {
    if ((int)threadIdx.x < o) red[threadIdx.x] += red[threadIdx.x + o];
    __syncthreads();
  }
  if (threadIdx.x == 0) out[(size_t)b * C + c] = red[0] / HW;
}

// tiny dense layer, leaky relu (leaky=1 -> identity)
__global__ void fc_kernel(const float* __restrict__ x, const float* __restrict__ w,
                          const float* __restrict__ b, float* __restrict__ y,
                          int rows, int In, int Out, float leaky)
{
  int idx = blockIdx.x * blockDim.x + threadIdx.x;
  if (idx >= rows * Out) return;
  int row = idx / Out, o = idx % Out;
  const float* xr = x + (size_t)row * In;
  const float* wr = w + (size_t)o * In;
  float acc = b ? b[o] : 0.f;
  for (int i = 0; i < In; i++) acc += xr[i] * wr[i];
  if (acc < 0.f) acc *= leaky;
  y[idx] = acc;
}

// Haar DWT: x (B,C,H,W) -> ll (batch stride ll_bs) + [lh|hl|hh] (batch stride h_bs)
__global__ void dwt2_kernel(const float* __restrict__ x, float* __restrict__ dll, size_t ll_bs,
                            float* __restrict__ dhigh, size_t h_bs, int B, int C, int H, int W)
{
  int H2 = H / 2, W2 = W / 2, HW2 = H2 * W2;
  int n = B * C * HW2;
  int idx = blockIdx.x * blockDim.x + threadIdx.x;
  if (idx >= n) return;
  int p = idx % HW2; int t = idx / HW2; int c = t % C; int b = t / C;
  int i = p / W2, j = p % W2;
  const float* xp = x + ((size_t)b * C + c) * (size_t)H * W;
  float a = xp[(2 * i) * W + 2 * j],     e = xp[(2 * i) * W + 2 * j + 1];
  float f = xp[(2 * i + 1) * W + 2 * j], h = xp[(2 * i + 1) * W + 2 * j + 1];
  dll[(size_t)b * ll_bs + (size_t)c * HW2 + p] = 0.25f * (a + e + f + h);
  float* hb = dhigh + (size_t)b * h_bs + p;
  hb[(size_t)(0 * C + c) * HW2] = 0.25f * (a + e - f - h);
  hb[(size_t)(1 * C + c) * HW2] = 0.25f * (a - e + f - h);
  hb[(size_t)(2 * C + c) * HW2] = 0.25f * (a - e - f + h);
}

// inverse Haar DWT: ll (B,C,H2,W2), high (B,3C,H2,W2) -> out (B,C,2H2,2W2)
__global__ void idwt2_kernel(const float* __restrict__ ll, const float* __restrict__ high,
                             float* __restrict__ out, int B, int C, int H2, int W2)
{
  int HW2 = H2 * W2;
  int n = B * C * HW2;
  int idx = blockIdx.x * blockDim.x + threadIdx.x;
  if (idx >= n) return;
  int p = idx % HW2; int t = idx / HW2; int c = t % C; int b = t / C;
  int i = p / W2, j = p % W2;
  float llv = ll[idx];
  const float* hb = high + ((size_t)b * 3 * C + c) * HW2 + p;
  float lh = hb[0], hl = hb[(size_t)C * HW2], hh = hb[2 * (size_t)C * HW2];
  int H = 2 * H2, W = 2 * W2;
  float* ob = out + ((size_t)b * C + c) * (size_t)H * W;
  ob[(2 * i) * W + 2 * j]         = llv + lh + hl + hh;
  ob[(2 * i) * W + 2 * j + 1]     = llv + lh - hl - hh;
  ob[(2 * i + 1) * W + 2 * j]     = llv - lh + hl - hh;
  ob[(2 * i + 1) * W + 2 * j + 1] = llv - lh - hl + hh;
}

// x (B,C,L) -> out (B,L,C), optional broadcast add z(B,C)
__global__ void nchw_to_blc_kernel(const float* __restrict__ x, const float* __restrict__ z,
                                   float* __restrict__ out, int B, int C, int L)
{
  int n = B * C * L;
  int idx = blockIdx.x * blockDim.x + threadIdx.x;
  if (idx >= n) return;
  int l = idx % L; int t = idx / L; int c = t % C; int b = t / C;
  float v = x[idx];
  if (z) v += z[(size_t)b * C + c];
  out[((size_t)b * L + l) * C + c] = v;
}

// dst (B,C,L) += y (B,L,C)
__global__ void blc_add_nchw_kernel(const float* __restrict__ y, float* __restrict__ dst,
                                    int B, int C, int L)
{
  int n = B * C * L;
  int idx = blockIdx.x * blockDim.x + threadIdx.x;
  if (idx >= n) return;
  int l = idx % L; int t = idx / L; int c = t % C; int b = t / C;
  dst[idx] += y[((size_t)b * L + l) * C + c];
}

// strided column-block copy (pointers pre-offset to the target column)
__global__ void copy_cols_kernel(const float* __restrict__ src, int s_src,
                                 float* __restrict__ dst, int s_dst, int rows, int C)
{
  int n = rows * C;
  int idx = blockIdx.x * blockDim.x + threadIdx.x;
  if (idx >= n) return;
  int r = idx / C, c = idx % C;
  dst[(size_t)r * s_dst + c] = src[(size_t)r * s_src + c];
}

// causal depthwise conv (kernel 4, left pad 3) + SiLU, dual-layout output
__global__ void dwconv_silu_kernel(const float* __restrict__ xin, int s_in,
                                   const float* __restrict__ cw, const float* __restrict__ cb,
                                   float* __restrict__ xc_cl, float* __restrict__ xc_lc,
                                   int B, int C, int L)
{
  int n = B * C * L;
  int idx = blockIdx.x * blockDim.x + threadIdx.x;
  if (idx >= n) return;
  int l = idx % L; int t = idx / L; int c = t % C; int b = t / C;
  const float* base = xin + (size_t)b * L * s_in + c;
  float acc = cb[c];
#pragma unroll
  for (int k = 0; k < 4; k++) {
    int ls = l - 3 + k;
    if (ls >= 0) acc += cw[c * 4 + k] * base[(size_t)ls * s_in];
  }
  acc = acc / (1.f + __expf(-acc));   // silu
  xc_cl[idx] = acc;
  xc_lc[((size_t)b * L + l) * C + c] = acc;
}

// LayerNorm over last dim: one wave32 per row
__global__ void layernorm_kernel(const float* __restrict__ x, const float* __restrict__ gam,
                                 const float* __restrict__ bet, float* __restrict__ y,
                                 int rows, int C)
{
  int wid  = (blockIdx.x * blockDim.x + threadIdx.x) >> 5;
  int lane = threadIdx.x & 31;
  if (wid >= rows) return;
  const float* xr = x + (size_t)wid * C;
  float s = 0.f, s2 = 0.f;
  for (int c = lane; c < C; c += 32) { float v = xr[c]; s += v; s2 += v * v; }
  for (int o = 16; o > 0; o >>= 1) { s += __shfl_xor(s, o); s2 += __shfl_xor(s2, o); }
  float m = s / C;
  float inv = rsqrtf(s2 / C - m * m + 1e-5f);
  float* yr = y + (size_t)wid * C;
  for (int c = lane; c < C; c += 32) yr[c] = (xr[c] - m) * inv * gam[c] + bet[c];
}

// ---------------------------------------------------------------------------
// Chunked selective scan (3 passes).  Recurrence h <- exp(d*A)*h + d*u*B,
// y = <h,C> + D*u, gated by silu(z).  Chunk products collapse via
// prod exp(d_t*A) = exp(A * sum d_t).
// ---------------------------------------------------------------------------
__global__ void scan_pass1(const float* __restrict__ u, const float* __restrict__ dlt, int s_dt,
                           const float* __restrict__ dt_b, const float* __restrict__ A_log,
                           const float* __restrict__ bc, int s_bc, int off_b,
                           float* __restrict__ sumd, float* __restrict__ hloc,
                           int B, int C, int L, int CH, int NC)
{
  int idx = blockIdx.x * blockDim.x + threadIdx.x;
  if (idx >= B * C * NC) return;
  int ch = idx % NC; int t0 = idx / NC; int c = t0 % C; int b = t0 / C;
  float A[16], h[16];
#pragma unroll
  for (int n = 0; n < 16; n++) { A[n] = -__expf(A_log[c * 16 + n]); h[n] = 0.f; }
  float db = dt_b[c];
  const float* up = u + ((size_t)b * C + c) * L + ch * CH;
  float sd = 0.f;
  for (int tt = 0; tt < CH; tt++) {
    size_t rl = (size_t)b * L + ch * CH + tt;
    float dv = dlt[rl * s_dt + c] + db;
    dv = (dv > 20.f) ? dv : __logf(1.f + __expf(dv));
    sd += dv;
    float du = dv * up[tt];
    const float* bp = bc + rl * s_bc + off_b;
#pragma unroll
    for (int n = 0; n < 16; n++) h[n] = __expf(dv * A[n]) * h[n] + du * bp[n];
  }
  sumd[idx] = sd;
#pragma unroll
  for (int n = 0; n < 16; n++) hloc[(size_t)idx * 16 + n] = h[n];
}

__global__ void scan_pass2(const float* __restrict__ A_log, const float* __restrict__ sumd,
                           const float* __restrict__ hloc, float* __restrict__ hstart,
                           int B, int C, int NC)
{
  int idx = blockIdx.x * blockDim.x + threadIdx.x;
  if (idx >= B * C) return;
  int c = idx % C;
  float A[16], h[16];
#pragma unroll
  for (int n = 0; n < 16; n++) { A[n] = -__expf(A_log[c * 16 + n]); h[n] = 0.f; }
  size_t base = (size_t)idx * NC;
  for (int ch = 0; ch < NC; ch++) {
    float sd = sumd[base + ch];
#pragma unroll
    for (int n = 0; n < 16; n++) {
      hstart[(base + ch) * 16 + n] = h[n];
      h[n] = __expf(A[n] * sd) * h[n] + hloc[(base + ch) * 16 + n];
    }
  }
}

__global__ void scan_pass3(const float* __restrict__ u, const float* __restrict__ dlt, int s_dt,
                           const float* __restrict__ dt_b, const float* __restrict__ A_log,
                           const float* __restrict__ bc, int s_bc, int off_b,
                           const float* __restrict__ Dp,
                           const float* __restrict__ zrow, int s_z, int off_z,
                           const float* __restrict__ hstart, float* __restrict__ y,
                           int B, int C, int L, int CH, int NC)
{
  int idx = blockIdx.x * blockDim.x + threadIdx.x;
  if (idx >= B * C * NC) return;
  int ch = idx % NC; int t0 = idx / NC; int c = t0 % C; int b = t0 / C;
  float A[16], h[16];
#pragma unroll
  for (int n = 0; n < 16; n++) {
    A[n] = -__expf(A_log[c * 16 + n]);
    h[n] = hstart[(size_t)idx * 16 + n];
  }
  float db = dt_b[c], Dc = Dp[c];
  const float* up = u + ((size_t)b * C + c) * L + ch * CH;
  for (int tt = 0; tt < CH; tt++) {
    size_t rl = (size_t)b * L + ch * CH + tt;
    float dv = dlt[rl * s_dt + c] + db;
    dv = (dv > 20.f) ? dv : __logf(1.f + __expf(dv));
    float uv = up[tt];
    float du = dv * uv;
    const float* bp = bc + rl * s_bc + off_b;
    float yv = 0.f;
#pragma unroll
    for (int n = 0; n < 16; n++) {
      h[n] = __expf(dv * A[n]) * h[n] + du * bp[n];
      yv += h[n] * bp[16 + n];
    }
    float zv = zrow[rl * s_z + off_z + c];
    float sig = zv / (1.f + __expf(-zv));
    y[rl * C + c] = (yv + Dc * uv) * sig;
  }
}

// ---------------------------------------------------------------------------
// ADR modulation
// ---------------------------------------------------------------------------
__global__ void adr_stats_kernel(const float* __restrict__ x, float* __restrict__ feat,
                                 int C, int HW)
{
  int c = blockIdx.x, b = blockIdx.y;
  const float* p = x + ((size_t)b * C + c) * HW;
  float s = 0.f, s2 = 0.f, sa = 0.f;
  for (int i = threadIdx.x; i < HW; i += blockDim.x) {
    float v = p[i]; s += v; s2 += v * v; sa += fabsf(v);
  }
  __shared__ float r0[256], r1[256], r2[256];
  r0[threadIdx.x] = s; r1[threadIdx.x] = s2; r2[threadIdx.x] = sa;
  __syncthreads();
  for (int o = 128; o > 0; o >>= 1) {
    if ((int)threadIdx.x < o) {
      r0[threadIdx.x] += r0[threadIdx.x + o];
      r1[threadIdx.x] += r1[threadIdx.x + o];
      r2[threadIdx.x] += r2[threadIdx.x + o];
    }
    __syncthreads();
  }
  if (threadIdx.x == 0) {
    float mean = r0[0] / HW;
    float var  = (r1[0] - r0[0] * r0[0] / HW) / (HW - 1);
    feat[((size_t)b * 3 + 0) * C + c] = r2[0] / HW;
    feat[((size_t)b * 3 + 1) * C + c] = mean;
    feat[((size_t)b * 3 + 2) * C + c] = sqrtf(var + 1e-6f);
  }
}

__global__ void adr_conv_kernel(const float* __restrict__ feat, const float* __restrict__ w,
                                float* __restrict__ scale, int B, int C)
{
  int idx = blockIdx.x * blockDim.x + threadIdx.x;
  if (idx >= B * C) return;
  int b = idx / C, c = idx % C;
  float acc = 0.f;
  for (int i = 0; i < 3; i++)
    for (int k = 0; k < 3; k++) {
      int cc = c - 1 + k;
      if (cc >= 0 && cc < C) acc += w[i * 3 + k] * feat[((size_t)b * 3 + i) * C + cc];
    }
  scale[idx] = 1.f / (1.f + __expf(-acc));
}

__global__ void scale_channels_kernel(float* __restrict__ x, const float* __restrict__ scale,
                                      int B, int C, int HW)
{
  int n = B * C * HW;
  int idx = blockIdx.x * blockDim.x + threadIdx.x;
  if (idx >= n) return;
  x[idx] *= scale[idx / HW];
}

// ---------------------------------------------------------------------------
// Host orchestration
// ---------------------------------------------------------------------------
static inline int cdiv(int a, int b) { return (a + b - 1) / b; }

static void gemm(hipStream_t s, const float* X, int lda, const _Float16* W16, const float* bias,
                 float* Y, int ldc, int M, int N, int K, int act)
{
  dim3 grid(cdiv(M, GTM), cdiv(N, GTN));
  wmma_gemm_kernel<<<grid, 128, 0, s>>>(X, lda, W16, bias, Y, ldc, M, N, K, act);
}

static void launch_scan(hipStream_t st, const float* u, const float* dlt, int s_dt,
                        const float* dt_b, const float* A_log, const float* bc, int s_bc,
                        int off_b, const float* Dp, const float* zrow, int s_z, int off_z,
                        float* y, int B_, int C_, int L_,
                        float* sumd, float* hloc, float* hstart)
{
  const int CH = 64, NC = L_ / CH;
  int n1 = B_ * C_ * NC;
  scan_pass1<<<cdiv(n1, 256), 256, 0, st>>>(u, dlt, s_dt, dt_b, A_log, bc, s_bc, off_b,
                                            sumd, hloc, B_, C_, L_, CH, NC);
  int n2 = B_ * C_;
  scan_pass2<<<cdiv(n2, 256), 256, 0, st>>>(A_log, sumd, hloc, hstart, B_, C_, NC);
  scan_pass3<<<cdiv(n1, 256), 256, 0, st>>>(u, dlt, s_dt, dt_b, A_log, bc, s_bc, off_b,
                                            Dp, zrow, s_z, off_z, hstart, y, B_, C_, L_, CH, NC);
}

extern "C" void kernel_launch(void* const* d_in, const int* in_sizes, int n_in,
                              void* d_out, int out_size, void* d_ws, size_t ws_size,
                              hipStream_t stream)
{
  (void)in_sizes; (void)out_size; (void)ws_size;
  if (n_in < 131) return;

  const int B = 8, H0 = 128, W0 = 128, D = 64;
  const int HW0 = H0 * W0;
  const int L = 64 * 64;          // 4096 (post-DWT spatial)
  const int M = B * L;            // 32768 GEMM rows

  auto F = [&](int i) { return (const float*)d_in[i]; };
  const float* ms  = F(0);
  const float* pan = F(1);

  // ---- workspace carve-out ----
  float* ws = (float*)d_ws;
  size_t off = 0;
  auto alloc = [&](size_t n) { float* p = ws + off; off += (n + 63) & ~(size_t)63; return p; };

  float* f_ms   = alloc((size_t)B * D * HW0);
  float* m_ll   = alloc((size_t)B * D * L);
  float* m_high = alloc((size_t)B * 3 * D * L);
  float* qf     = alloc((size_t)M * 192);
  float* kf     = alloc((size_t)M * 192);
  float* zb     = alloc(B * D);
  float* gpool  = alloc(B * D);
  float* gfc    = alloc(B * D * 4);
  float* feat   = alloc(B * 3 * D);
  float* scl    = alloc(B * D);
  float* fbuf   = alloc((size_t)B * D * HW0);
  float* sumd   = alloc((size_t)B * 384 * 64);
  float* hloc   = alloc((size_t)B * 384 * 64 * 16);
  float* hstart = alloc((size_t)B * 384 * 64 * 16);
  // block-loop temporaries (largest consumers)
  float* t_xf    = alloc((size_t)M * 192);
  float* t_in    = alloc((size_t)M * 384);
  float* t_xc_cl = alloc((size_t)M * 384);
  float* t_xc_lc = alloc((size_t)M * 384);
  float* t_ln    = alloc((size_t)M * 192);
  float* t_kp    = alloc((size_t)M * 384);
  float* t_qp    = alloc((size_t)M * 384);
  float* t_cat   = alloc((size_t)M * 768);
  float* t_xdbl  = alloc((size_t)M * 44);
  float* t_gate  = alloc((size_t)M * 384);
  float* t_delta = alloc((size_t)M * 384);
  float* t_y     = alloc((size_t)M * 384);
  float* t_yout  = alloc((size_t)M * 192);
  // prep-phase aliases (lifetimes end before the block loop)
  float* g1     = t_in;      // B*32*HW0
  float* g2     = t_xc_cl;   // B*64*HW0
  float* f_pan  = t_xc_lc;   // B*64*HW0
  float* pll    = t_kp;      // B*64*L
  float* pll_lc = t_qp;      // M*64
  float* p_high = t_gate;    // B*192*L

  // ---- f16 weight pool + conversion (per launch; deterministic) ----
  _Float16* h16pool = (_Float16*)(ws + off);
  size_t h16off = 0;
  _Float16* W16[160] = {};
  auto cv = [&](int i, int n) {
    _Float16* d = h16pool + h16off;
    h16off += ((size_t)n + 63) & ~(size_t)63;
    f32_to_f16_kernel<<<cdiv(n, 256), 256, 0, stream>>>(F(i), d, n);
    W16[i] = d;
  };
  cv(14, 192 * 64);                               // kexp
  for (int blk = 0; blk < 4; blk++) {
    int s = 19 + blk * 28, t = s + 9;
    cv(s + 7, 256 * 64);  cv(s + 8, 36 * 128);    // vim in_w, x_w
    cv(s + 2, 128 * 4);   cv(s + 6, 64 * 128);    // vim dt_w, out_w
    cv(t + 7, 384 * 192); cv(t + 10, 384 * 192);  // pid in_w, k_w
    cv(t + 14, 384 * 192); cv(t + 16, 44 * 768);  // pid q_w, dtbc_w
    cv(t + 17, 384 * 768); cv(t + 2, 384 * 12);   // pid gate_w, dt_w
    cv(t + 6, 192 * 384);                         // pid out_w
  }

  // ---- GSA: z = MLP(GAP(conv(conv(ms)))) ----
  conv3x3_kernel<<<dim3(32, B), 256, 0, stream>>>(ms, F(6), F(7), nullptr, g1, B, 8, 32, H0, W0, 0.1f);
  conv3x3_kernel<<<dim3(64, B), 256, 0, stream>>>(g1, F(8), F(9), nullptr, g2, B, 32, 64, H0, W0, 0.1f);
  gap_kernel<<<dim3(64, B), 256, 0, stream>>>(g2, gpool, HW0);
  fc_kernel<<<cdiv(B * 256, 128), 128, 0, stream>>>(gpool, F(10), F(11), gfc, B, 64, 256, 0.1f);
  fc_kernel<<<cdiv(B * 64, 128), 128, 0, stream>>>(gfc, F(12), F(13), zb, B, 256, 64, 1.0f);

  // ---- stem convs + DWT ----
  conv3x3_kernel<<<dim3(64, B), 256, 0, stream>>>(ms, F(2), F(3), nullptr, f_ms, B, 8, 64, H0, W0, 1.0f);
  conv3x3_kernel<<<dim3(64, B), 256, 0, stream>>>(pan, F(4), F(5), nullptr, f_pan, B, 1, 64, H0, W0, 1.0f);
  {
    int n = B * D * L;
    dwt2_kernel<<<cdiv(n, 256), 256, 0, stream>>>(f_ms, m_ll, (size_t)D * L, m_high, (size_t)3 * D * L, B, D, H0, W0);
    dwt2_kernel<<<cdiv(n, 256), 256, 0, stream>>>(f_pan, pll, (size_t)D * L, p_high, (size_t)3 * D * L, B, D, H0, W0);
  }
  nchw_to_blc_kernel<<<cdiv(B * 192 * L, 256), 256, 0, stream>>>(p_high, nullptr, qf, B, 192, L);
  nchw_to_blc_kernel<<<cdiv(B * 64 * L, 256), 256, 0, stream>>>(pll, nullptr, pll_lc, B, 64, L);
  // K = 1x1 conv(p_ll) as GEMM, written directly in (B,L,192)
  gemm(stream, pll_lc, 64, W16[14], F(15), kf, 192, M, 192, 64, 0);

  // ---- 4 blocks ----
  for (int blk = 0; blk < 4; blk++) {
    int s = 19 + blk * 28;  // vim (sss) params
    int t = s + 9;          // pid (sts) params

    // -------- vim on m_ll: dim 64, din 128, dr 4 --------
    nchw_to_blc_kernel<<<cdiv(B * 64 * L, 256), 256, 0, stream>>>(m_ll, zb, t_xf, B, 64, L);
    gemm(stream, t_xf, 64, W16[s + 7], nullptr, t_in, 256, M, 256, 64, 0);
    dwconv_silu_kernel<<<cdiv(B * 128 * L, 256), 256, 0, stream>>>(t_in, 256, F(s + 0), F(s + 1),
                                                                   t_xc_cl, t_xc_lc, B, 128, L);
    gemm(stream, t_xc_lc, 128, W16[s + 8], nullptr, t_xdbl, 36, M, 36, 128, 0);
    gemm(stream, t_xdbl, 36, W16[s + 2], F(s + 3), t_delta, 128, M, 128, 4, 0);
    launch_scan(stream, t_xc_cl, t_delta, 128, F(s + 3), F(s + 4), t_xdbl, 36, 4,
                F(s + 5), t_in, 256, 128, t_y, B, 128, L, sumd, hloc, hstart);
    gemm(stream, t_y, 128, W16[s + 6], nullptr, t_yout, 64, M, 64, 128, 0);
    blc_add_nchw_kernel<<<cdiv(B * 64 * L, 256), 256, 0, stream>>>(t_yout, m_ll, B, 64, L);

    // -------- pid on m_high: dim 192, din 384, dr 12 --------
    nchw_to_blc_kernel<<<cdiv(B * 192 * L, 256), 256, 0, stream>>>(m_high, nullptr, t_xf, B, 192, L);
    gemm(stream, t_xf, 192, W16[t + 7], nullptr, t_in, 384, M, 384, 192, 0);
    dwconv_silu_kernel<<<cdiv(B * 384 * L, 256), 256, 0, stream>>>(t_in, 384, F(t + 0), F(t + 1),
                                                                   t_xc_cl, t_xc_lc, B, 384, L);
    layernorm_kernel<<<cdiv(M * 32, 256), 256, 0, stream>>>(kf, F(t + 8), F(t + 9), t_ln, M, 192);
    gemm(stream, t_ln, 192, W16[t + 10], F(t + 11), t_kp, 384, M, 384, 192, 1);
    layernorm_kernel<<<cdiv(M * 32, 256), 256, 0, stream>>>(qf, F(t + 12), F(t + 13), t_ln, M, 192);
    gemm(stream, t_ln, 192, W16[t + 14], F(t + 15), t_qp, 384, M, 384, 192, 1);
    copy_cols_kernel<<<cdiv(M * 384, 256), 256, 0, stream>>>(t_xc_lc, 384, t_cat, 768, M, 384);
    copy_cols_kernel<<<cdiv(M * 384, 256), 256, 0, stream>>>(t_kp, 384, t_cat + 384, 768, M, 384);
    gemm(stream, t_cat, 768, W16[t + 16], nullptr, t_xdbl, 44, M, 44, 768, 0);
    copy_cols_kernel<<<cdiv(M * 384, 256), 256, 0, stream>>>(t_qp, 384, t_cat + 384, 768, M, 384);
    gemm(stream, t_cat, 768, W16[t + 17], F(t + 18), t_gate, 384, M, 384, 768, 0);
    gemm(stream, t_xdbl, 44, W16[t + 2], F(t + 3), t_delta, 384, M, 384, 12, 0);
    launch_scan(stream, t_xc_cl, t_delta, 384, F(t + 3), F(t + 4), t_xdbl, 44, 12,
                F(t + 5), t_gate, 384, 0, t_y, B, 384, L, sumd, hloc, hstart);
    gemm(stream, t_y, 384, W16[t + 6], nullptr, t_yout, 192, M, 192, 384, 0);
    blc_add_nchw_kernel<<<cdiv(B * 192 * L, 256), 256, 0, stream>>>(t_yout, m_high, B, 192, L);
  }

  // ---- IDWT + ADR + tail ----
  idwt2_kernel<<<cdiv(B * 64 * L, 256), 256, 0, stream>>>(m_ll, m_high, fbuf, B, 64, 64, 64);
  adr_stats_kernel<<<dim3(64, B), 256, 0, stream>>>(fbuf, feat, 64, HW0);
  adr_conv_kernel<<<cdiv(B * 64, 128), 128, 0, stream>>>(feat, F(16), scl, B, 64);
  scale_channels_kernel<<<cdiv(B * 64 * HW0, 256), 256, 0, stream>>>(fbuf, scl, B, 64, HW0);
  conv3x3_kernel<<<dim3(8, B), 256, 0, stream>>>(fbuf, F(17), F(18), ms, (float*)d_out,
                                                 B, 64, 8, H0, W0, 1.0f);
}
